// CNV_7825430413870
// MI455X (gfx1250) — 8-bit matrix operand order repaired and checked by reading, compiled, not run on hardware
//
#include <hip/hip_runtime.h>
#include <stdint.h>
#include <math.h>

typedef __attribute__((ext_vector_type(8))) int v8i;

#define EPSF 1e-4f
#define NIMG 1024

__global__ __launch_bounds__(256) void absmean_kernel(const float* __restrict__ w, int n,
                                                      float* __restrict__ out) {
  __shared__ float red[256];
  float s = 0.f;
  for (int i = threadIdx.x; i < n; i += 256) s += fabsf(w[i]);
  red[threadIdx.x] = s;
  __syncthreads();
  for (int off = 128; off > 0; off >>= 1) {
    if ((int)threadIdx.x < off) red[threadIdx.x] += red[threadIdx.x + off];
    __syncthreads();
  }
  if (threadIdx.x == 0) *out = red[0] / (float)n;
}

__global__ __launch_bounds__(256) void pack_weights_kernel(const float* __restrict__ w,
                                                           int8_t* __restrict__ packed,
                                                           int CIN, int KHW, int Klog,
                                                           int KC, int ntiles) {
  int total = ntiles * KC * 256;
  int idx = blockIdx.x * 256 + threadIdx.x;
  if (idx >= total) return;
  int d = idx & 7;
  int lane = (idx >> 3) & 31;
  int ntkc = idx >> 8;
  int kc = ntkc % KC;
  int nt = ntkc / KC;
  int col = lane & 15, kh = lane >> 4;
  int cout = nt * 16 + col;
  int kbase = kc * 64 + (d >> 1) * 16 + kh * 8 + (d & 1) * 4;
  uint32_t dw = 0;
  for (int j = 0; j < 4; ++j) {
    int k = kbase + j;
    int v = 0;
    if (k < Klog) {
      int pos = k / CIN, ci = k - pos * CIN;
      float f = w[(cout * CIN + ci) * KHW + pos];
      v = (f > 0.f) ? 1 : ((f < 0.f) ? -1 : 0);
    }
    dw |= ((uint32_t)(uint8_t)(int8_t)v) << (8 * j);
  }
  ((uint32_t*)packed)[idx] = dw;
}

__global__ __launch_bounds__(256) void quant_input_kernel(const float* __restrict__ x,
                                                          int8_t* __restrict__ out) {
  int idx = blockIdx.x * 256 + threadIdx.x;
  const int total = NIMG * 32 * 32 * 3;
  if (idx >= total) return;
  int c = idx % 3;
  int t = idx / 3;
  int wc = t % 32; t /= 32;
  int h = t % 32;
  int n = t / 32;
  float v = x[((n * 3 + c) * 32 + h) * 32 + wc];
  float q = rintf(256.f * v - 128.f);
  q = fminf(fmaxf(q, -128.f), 127.f);
  out[idx] = (int8_t)(int)q;
}

#define SMEM_A_BYTES (16 * 2304)

__global__ __launch_bounds__(128) void bgemm_conv_kernel(
    const int8_t* __restrict__ act_in, int8_t* __restrict__ act_out,
    const int8_t* __restrict__ packedB, const float* __restrict__ wscale,
    const float* __restrict__ bn_g, const float* __restrict__ bn_b,
    const float* __restrict__ bn_m, const float* __restrict__ bn_v,
    float in_scale,
    int IH, int IW, int CIN, int OH, int OW, int COUT,
    int KH, int KW, int Klog, int Kpad) {
  __shared__ int8_t smA[SMEM_A_BYTES] __attribute__((aligned(16)));
  const int tid = threadIdx.x;
  const int lane = tid & 31;
  const int wave = tid >> 5;
  const int mtile = blockIdx.x;
  const int ntile = blockIdx.y * 4 + wave;
  const int KC = Kpad >> 6;
  const int OHW = OH * OW;
  const int segLen = KW * CIN;

  if ((CIN & 3) == 0) {
    const int segD = segLen >> 2;
    const int rowD = KH * segD;
    for (int t = tid; t < 16 * rowD; t += 128) {
      int r = t / rowD;
      int rem = t - r * rowD;
      int ky = rem / segD;
      int od = rem - ky * segD;
      int m = mtile * 16 + r;
      int n = m / OHW;
      int rem2 = m - n * OHW;
      int oh = rem2 / OW, ow = rem2 - (rem2 / OW) * OW;
      int gofs = ((n * IH + oh + ky) * IW + ow) * CIN + (od << 2);
      *(int*)(smA + r * Kpad + ky * segLen + (od << 2)) = *(const int*)(act_in + gofs);
    }
    const int padD = (Kpad - Klog) >> 2;
    for (int t = tid; t < 16 * padD; t += 128) {
      int r = t / padD;
      int od = t - r * padD;
      *(int*)(smA + r * Kpad + Klog + (od << 2)) = 0;
    }
  } else {
    const int rowB = KH * segLen;
    for (int t = tid; t < 16 * rowB; t += 128) {
      int r = t / rowB;
      int rem = t - r * rowB;
      int ky = rem / segLen;
      int ob = rem - ky * segLen;
      int m = mtile * 16 + r;
      int n = m / OHW;
      int rem2 = m - n * OHW;
      int oh = rem2 / OW, ow = rem2 - (rem2 / OW) * OW;
      smA[r * Kpad + ky * segLen + ob] =
          act_in[((n * IH + oh + ky) * IW + ow) * CIN + ob];
    }
    const int per = Kpad - Klog;
    for (int t = tid; t < 16 * per; t += 128) {
      int r = t / per;
      int ob = t - r * per;
      smA[r * Kpad + Klog + ob] = 0;
    }
  }
  __syncthreads();

  const int row = lane & 15;
  const int kh16 = lane >> 4;
  v8i acc = {0, 0, 0, 0, 0, 0, 0, 0};
  const int8_t* bptr = packedB + (size_t)ntile * KC * 1024 + (size_t)lane * 32;
  for (int kc = 0; kc < KC; ++kc) {
    v8i a;
    const int abase = row * Kpad + (kc << 6) + (kh16 << 3);
#pragma unroll
    for (int g = 0; g < 4; ++g) {
      int2 t2 = *(const int2*)(smA + abase + (g << 4));
      a[2 * g] = t2.x;
      a[2 * g + 1] = t2.y;
    }
    const int4* bp = (const int4*)(bptr + (size_t)kc * 1024);
    int4 b0 = bp[0];
    int4 b1 = bp[1];
    v8i b = {b0.x, b0.y, b0.z, b0.w, b1.x, b1.y, b1.z, b1.w};
    acc = __builtin_amdgcn_wmma_i32_16x16x64_iu8(true, a, true, b, acc, false, false);
  }

  const int cout = ntile * 16 + row;
  const float ig = bn_g[cout] * (1.0f / sqrtf(bn_v[cout] + EPSF));
  const float alpha = in_scale * (*wscale) * ig;
  const float beta = bn_b[cout] - bn_m[cout] * ig;
#pragma unroll
  for (int d = 0; d < 8; ++d) {
    int m = mtile * 16 + d + (kh16 << 3);
    int n = m / OHW;
    int rem2 = m - n * OHW;
    int oh = rem2 / OW, ow = rem2 - (rem2 / OW) * OW;
    float z = (float)acc[d] * alpha + beta;
    act_out[((size_t)(n * OH + oh) * OW + ow) * COUT + cout] =
        (z >= 0.f) ? (int8_t)1 : (int8_t)-1;
  }
}

__global__ __launch_bounds__(256) void maxpool_kernel(const int8_t* __restrict__ in,
                                                      int8_t* __restrict__ out,
                                                      int IH, int IW, int C) {
  int OH = IH >> 1, OW = IW >> 1;
  size_t total = (size_t)NIMG * OH * OW * C;
  size_t idx = (size_t)blockIdx.x * 256 + threadIdx.x;
  if (idx >= total) return;
  int c = (int)(idx % C);
  size_t t = idx / C;
  int ow = (int)(t % OW); t /= OW;
  int oh = (int)(t % OH);
  int n = (int)(t / OH);
  const int8_t* base = in + (((size_t)n * IH + 2 * oh) * IW + 2 * ow) * C + c;
  int8_t v0 = base[0], v1 = base[C];
  int8_t v2 = base[(size_t)IW * C], v3 = base[(size_t)IW * C + C];
  int8_t m01 = v0 > v1 ? v0 : v1;
  int8_t m23 = v2 > v3 ? v2 : v3;
  out[idx] = m01 > m23 ? m01 : m23;
}

__global__ __launch_bounds__(256) void head_kernel(const int8_t* __restrict__ act,
                                                   const float* __restrict__ w,
                                                   const float* __restrict__ wscale,
                                                   const float* __restrict__ ln_g,
                                                   const float* __restrict__ ln_b,
                                                   float* __restrict__ out) {
  int n = blockIdx.x * 256 + threadIdx.x;
  if (n >= NIMG) return;
  int acc[10];
#pragma unroll
  for (int c = 0; c < 10; ++c) acc[c] = 0;
  for (int k = 0; k < 512; ++k) {
    int a = (int)act[n * 512 + k];
#pragma unroll
    for (int c = 0; c < 10; ++c) {
      float f = w[c * 512 + k];
      int sg = (f > 0.f) ? 1 : ((f < 0.f) ? -1 : 0);
      acc[c] += a * sg;
    }
  }
  const float s = *wscale;
  float y[10], mu = 0.f;
#pragma unroll
  for (int c = 0; c < 10; ++c) {
    y[c] = (float)acc[c] * s;
    mu += y[c];
  }
  mu *= 0.1f;
  float var = 0.f;
#pragma unroll
  for (int c = 0; c < 10; ++c) {
    float d = y[c] - mu;
    var += d * d;
  }
  var *= 0.1f;
  float inv = 1.0f / sqrtf(var + EPSF);
#pragma unroll
  for (int c = 0; c < 10; ++c) out[n * 10 + c] = (y[c] - mu) * inv * ln_g[c] + ln_b[c];
}

extern "C" void kernel_launch(void* const* d_in, const int* in_sizes, int n_in,
                              void* d_out, int out_size, void* d_ws, size_t ws_size,
                              hipStream_t stream) {
  (void)in_sizes; (void)n_in; (void)out_size; (void)ws_size;
  const float* x = (const float*)d_in[0];
  auto CW = [&](int i) { return (const float*)d_in[1 + 5 * i]; };
  auto CG = [&](int i) { return (const float*)d_in[2 + 5 * i]; };
  auto CB = [&](int i) { return (const float*)d_in[3 + 5 * i]; };
  auto CM = [&](int i) { return (const float*)d_in[4 + 5 * i]; };
  auto CV = [&](int i) { return (const float*)d_in[5 + 5 * i]; };
  const float* fc0_w = (const float*)d_in[31];
  const float* fb0_g = (const float*)d_in[32];
  const float* fb0_b = (const float*)d_in[33];
  const float* fb0_m = (const float*)d_in[34];
  const float* fb0_v = (const float*)d_in[35];
  const float* fc1_w = (const float*)d_in[36];
  const float* fb1_g = (const float*)d_in[37];
  const float* fb1_b = (const float*)d_in[38];
  const float* fb1_m = (const float*)d_in[39];
  const float* fb1_v = (const float*)d_in[40];
  const float* wlast = (const float*)d_in[41];
  const float* ln_g  = (const float*)d_in[42];
  const float* ln_b  = (const float*)d_in[43];
  float* out = (float*)d_out;

  int8_t* ws8 = (int8_t*)d_ws;
  const size_t OFF_A = 0;
  const size_t OFF_B = 64ull << 20;
  const size_t OFF_PACK = 128ull << 20;
  int8_t* actA = ws8 + OFF_A;
  int8_t* actB = ws8 + OFF_B;
  int8_t* pk0 = ws8 + OFF_PACK;
  int8_t* pk1 = pk0 + 4096;
  int8_t* pk2 = pk1 + 36864;
  int8_t* pk3 = pk2 + 73728;
  int8_t* pk4 = pk3 + 147456;
  int8_t* pk5 = pk4 + 294912;
  int8_t* pkf0 = pk5 + 589824;
  int8_t* pkf1 = pkf0 + 131072;
  float* scales = (float*)(pkf1 + 262144 + 256);

  absmean_kernel<<<1, 256, 0, stream>>>(CW(0), 64 * 3 * 9, scales + 0);
  absmean_kernel<<<1, 256, 0, stream>>>(CW(1), 64 * 64 * 9, scales + 1);
  absmean_kernel<<<1, 256, 0, stream>>>(CW(2), 128 * 64 * 9, scales + 2);
  absmean_kernel<<<1, 256, 0, stream>>>(CW(3), 128 * 128 * 9, scales + 3);
  absmean_kernel<<<1, 256, 0, stream>>>(CW(4), 256 * 128 * 9, scales + 4);
  absmean_kernel<<<1, 256, 0, stream>>>(CW(5), 256 * 256 * 9, scales + 5);
  absmean_kernel<<<1, 256, 0, stream>>>(fc0_w, 512 * 256, scales + 6);
  absmean_kernel<<<1, 256, 0, stream>>>(fc1_w, 512 * 512, scales + 7);
  absmean_kernel<<<1, 256, 0, stream>>>(wlast, 10 * 512, scales + 8);

  auto packGrid = [](int ntiles, int KC) { return (ntiles * KC * 256 + 255) / 256; };
  pack_weights_kernel<<<packGrid(4, 1), 256, 0, stream>>>(CW(0), pk0, 3, 9, 27, 1, 4);
  pack_weights_kernel<<<packGrid(4, 9), 256, 0, stream>>>(CW(1), pk1, 64, 9, 576, 9, 4);
  pack_weights_kernel<<<packGrid(8, 9), 256, 0, stream>>>(CW(2), pk2, 64, 9, 576, 9, 8);
  pack_weights_kernel<<<packGrid(8, 18), 256, 0, stream>>>(CW(3), pk3, 128, 9, 1152, 18, 8);
  pack_weights_kernel<<<packGrid(16, 18), 256, 0, stream>>>(CW(4), pk4, 128, 9, 1152, 18, 16);
  pack_weights_kernel<<<packGrid(16, 36), 256, 0, stream>>>(CW(5), pk5, 256, 9, 2304, 36, 16);
  pack_weights_kernel<<<packGrid(32, 4), 256, 0, stream>>>(fc0_w, pkf0, 256, 1, 256, 4, 32);
  pack_weights_kernel<<<packGrid(32, 8), 256, 0, stream>>>(fc1_w, pkf1, 512, 1, 512, 8, 32);

  quant_input_kernel<<<(NIMG * 32 * 32 * 3 + 255) / 256, 256, 0, stream>>>(x, actA);

  const float IN_SCALE = 1.0f / 128.0f;
  bgemm_conv_kernel<<<dim3(57600, 1), 128, 0, stream>>>(
      actA, actB, pk0, scales + 0, CG(0), CB(0), CM(0), CV(0), IN_SCALE,
      32, 32, 3, 30, 30, 64, 3, 3, 27, 64);
  bgemm_conv_kernel<<<dim3(50176, 1), 128, 0, stream>>>(
      actB, actA, pk1, scales + 1, CG(1), CB(1), CM(1), CV(1), 1.0f,
      30, 30, 64, 28, 28, 64, 3, 3, 576, 576);
  maxpool_kernel<<<(NIMG * 14 * 14 * 64 + 255) / 256, 256, 0, stream>>>(actA, actB, 28, 28, 64);
  bgemm_conv_kernel<<<dim3(9216, 2), 128, 0, stream>>>(
      actB, actA, pk2, scales + 2, CG(2), CB(2), CM(2), CV(2), 1.0f,
      14, 14, 64, 12, 12, 128, 3, 3, 576, 576);
  bgemm_conv_kernel<<<dim3(6400, 2), 128, 0, stream>>>(
      actA, actB, pk3, scales + 3, CG(3), CB(3), CM(3), CV(3), 1.0f,
      12, 12, 128, 10, 10, 128, 3, 3, 1152, 1152);
  maxpool_kernel<<<(NIMG * 5 * 5 * 128 + 255) / 256, 256, 0, stream>>>(actB, actA, 10, 10, 128);
  bgemm_conv_kernel<<<dim3(576, 4), 128, 0, stream>>>(
      actA, actB, pk4, scales + 4, CG(4), CB(4), CM(4), CV(4), 1.0f,
      5, 5, 128, 3, 3, 256, 3, 3, 1152, 1152);
  bgemm_conv_kernel<<<dim3(64, 4), 128, 0, stream>>>(
      actB, actA, pk5, scales + 5, CG(5), CB(5), CM(5), CV(5), 1.0f,
      3, 3, 256, 1, 1, 256, 3, 3, 2304, 2304);
  bgemm_conv_kernel<<<dim3(64, 8), 128, 0, stream>>>(
      actA, actB, pkf0, scales + 6, fb0_g, fb0_b, fb0_m, fb0_v, 1.0f,
      1, 1, 256, 1, 1, 512, 1, 1, 256, 256);
  bgemm_conv_kernel<<<dim3(64, 8), 128, 0, stream>>>(
      actB, actA, pkf1, scales + 7, fb1_g, fb1_b, fb1_m, fb1_v, 1.0f,
      1, 1, 512, 1, 1, 512, 1, 1, 512, 512);
  head_kernel<<<4, 256, 0, stream>>>(actA, wlast, scales + 8, ln_g, ln_b, out);
}
